// WaveletKAN_44719199486013
// MI455X (gfx1250) — matrix operand order repaired and checked by reading, not compiled, not run on hardware
//
#include <hip/hip_runtime.h>
#include <hip/hip_bf16.h>

typedef float v8f __attribute__((ext_vector_type(8)));
typedef __bf16 v8bf __attribute__((ext_vector_type(8)));
typedef __bf16 v16bf __attribute__((ext_vector_type(16)));
typedef int i32x4 __attribute__((vector_size(16)));  // matches builtin's V4i

#define AS1 __attribute__((address_space(1)))
#define AS3 __attribute__((address_space(3)))

#define D_DIM 1024
#define S_DIM 512
#define B_DIM 32
#define L_DIM 3
#define N_ROWS (B_DIM * S_DIM)  // 16384

#if defined(__has_builtin)
#if __has_builtin(__builtin_amdgcn_global_load_async_to_lds_b128)
#define HAVE_ASYNC_LDS 1
#endif
#if __has_builtin(__builtin_amdgcn_s_wait_asynccnt)
#define HAVE_WAIT_ASYNC 1
#endif
#endif

__device__ __forceinline__ float silu_f(float x) {
    return x / (1.0f + __expf(-x));
}

__device__ __forceinline__ unsigned short f2bf(float f) {
    union { float f; unsigned int u; } v; v.f = f;
    unsigned int r = v.u + 0x7FFFu + ((v.u >> 16) & 1u);  // round-to-nearest-even
    return (unsigned short)(r >> 16);
}

__device__ __forceinline__ void wait_async_lds() {
#if defined(HAVE_WAIT_ASYNC)
    __builtin_amdgcn_s_wait_asynccnt(0);
#elif defined(HAVE_ASYNC_LDS)
    asm volatile("s_wait_asynccnt 0" ::: "memory");
#endif
}

// Stage one 128x32-bf16 tile of A and of B into LDS; 1 row (64 B) per thread.
__device__ __forceinline__ void stage_tile(const unsigned short* gA,
                                           const unsigned short* gB,
                                           unsigned short (*As)[40],
                                           unsigned short (*Bs)[40], int tid) {
#if defined(HAVE_ASYNC_LDS)
#pragma unroll
    for (int q = 0; q < 4; ++q) {
        __builtin_amdgcn_global_load_async_to_lds_b128(
            (AS1 i32x4*)(gA + q * 8), (AS3 i32x4*)(&As[tid][q * 8]), 0, 0);
        __builtin_amdgcn_global_load_async_to_lds_b128(
            (AS1 i32x4*)(gB + q * 8), (AS3 i32x4*)(&Bs[tid][q * 8]), 0, 0);
    }
#else
    // Fallback: VGPR bounce, all loads issued before stores (distinct regs).
    const uint4* ga = (const uint4*)gA;
    const uint4* gb = (const uint4*)gB;
    uint4 ra0 = ga[0], ra1 = ga[1], ra2 = ga[2], ra3 = ga[3];
    uint4 rb0 = gb[0], rb1 = gb[1], rb2 = gb[2], rb3 = gb[3];
    uint4* la = (uint4*)(&As[tid][0]);
    uint4* lb = (uint4*)(&Bs[tid][0]);
    la[0] = ra0; la[1] = ra1; la[2] = ra2; la[3] = ra3;
    lb[0] = rb0; lb[1] = rb1; lb[2] = rb2; lb[3] = rb3;
#endif
}

// ---------------------------------------------------------------------------
// fp32 -> bf16 weight conversion
// ---------------------------------------------------------------------------
__global__ void f32_to_bf16_kernel(const float* __restrict__ src,
                                   unsigned short* __restrict__ dst, int n) {
    int idx = blockIdx.x * blockDim.x + threadIdx.x;
    if (idx < n) dst[idx] = f2bf(src[idx]);
}

// ---------------------------------------------------------------------------
// silu(h) -> sA (bf16, N x D) ; db1 DWT pair -> sP (bf16, N/2 x 2D)
// ---------------------------------------------------------------------------
__global__ void prep_silu_dwt_kernel(const float* __restrict__ h,
                                     unsigned short* __restrict__ sA,
                                     unsigned short* __restrict__ sP,
                                     int halfRows) {
    int idx = blockIdx.x * blockDim.x + threadIdx.x;
    if (idx >= halfRows * D_DIM) return;
    int d = idx & (D_DIM - 1);
    int p = idx >> 10;
    size_t re = (size_t)(2 * p) * D_DIM + d;
    size_t ro = re + D_DIM;
    float he = h[re], ho = h[ro];
    sA[re] = f2bf(silu_f(he));
    sA[ro] = f2bf(silu_f(ho));
    const float c = 0.70710678118654752440f;
    size_t pb = (size_t)p * (2 * D_DIM) + d;
    sP[pb]          = f2bf((he + ho) * c);  // cA
    sP[pb + D_DIM]  = f2bf((he - ho) * c);  // cD
}

// ---------------------------------------------------------------------------
// C[M,N] = A[M,K] * W[N,K]^T   (bf16 in, fp32 out)
// Block: 128 threads = 4 waves (wave32). Block tile 128x128, wave tile 64x64,
// K-tile 32. Double-buffered LDS fed by async global->LDS copies (ASYNCcnt),
// WMMA f32_16x16x32_bf16 accumulation.
// Requires M%128==0, N%128==0, K%32==0 (true for all calls here).
// ---------------------------------------------------------------------------
__global__ __launch_bounds__(128) void gemm_bf16_nt_kernel(
    const unsigned short* __restrict__ A, const unsigned short* __restrict__ W,
    float* __restrict__ C, int M, int N, int K) {
    // +8 halves of pad (16 B) keeps rows 16-byte aligned and skews banks
    __shared__ unsigned short As[2][128][40];
    __shared__ unsigned short Bs[2][128][40];

    const int tid  = threadIdx.x;
    const int lane = tid & 31;
    const int wave = tid >> 5;
    const int wr = wave >> 1;          // wave row (0..1)
    const int wc = wave & 1;           // wave col (0..1)
    const int m0 = blockIdx.y * 128;
    const int n0 = blockIdx.x * 128;

    v8f acc[4][4] = {};

    const unsigned short* aRow = A + (size_t)(m0 + tid) * K;
    const unsigned short* bRow = W + (size_t)(n0 + tid) * K;

    int cur = 0;
    stage_tile(aRow, bRow, As[0], Bs[0], tid);

    for (int kt = 0; kt < K; kt += 32) {
        wait_async_lds();          // this wave's copies into buf 'cur' are done
        __syncthreads();           // all waves' copies visible

        if (kt + 32 < K) {         // overlap: stream next tile into other buf
            stage_tile(aRow + kt + 32, bRow + kt + 32, As[cur ^ 1], Bs[cur ^ 1], tid);
        }

        // ---- gather A fragments (16x32 bf16 per ISA layout) ----
        // lanes 0-15:  halves 0..7 -> K0..7,  halves 8..15 -> K16..23
        // lanes 16-31: halves 0..7 -> K8..15, halves 8..15 -> K24..31
        v16bf afr[4];
#pragma unroll
        for (int i = 0; i < 4; ++i) {
            const int m   = wr * 64 + i * 16 + (lane & 15);
            const int klo = (lane < 16) ? 0 : 8;
            v8bf lo = *(const v8bf*)(&As[cur][m][klo]);
            v8bf hi = *(const v8bf*)(&As[cur][m][klo + 16]);
#pragma unroll
            for (int e = 0; e < 8; ++e) { afr[i][e] = lo[e]; afr[i][e + 8] = hi[e]; }
        }

        // ---- B fragments (32x16): lanes 0-15 hold K0..7,K16..23, lanes 16-31 K8..15,K24..31
#pragma unroll
        for (int j = 0; j < 4; ++j) {
            const int n  = wc * 64 + j * 16 + (lane & 15);
            const int kb = (lane < 16) ? 0 : 8;
            v8bf lo = *(const v8bf*)(&Bs[cur][n][kb]);
            v8bf hi = *(const v8bf*)(&Bs[cur][n][kb + 16]);
            v16bf bfr;
#pragma unroll
            for (int e = 0; e < 8; ++e) { bfr[e] = lo[e]; bfr[e + 8] = hi[e]; }
#pragma unroll
            for (int i = 0; i < 4; ++i) {
                acc[i][j] = __builtin_amdgcn_wmma_f32_16x16x32_bf16(
                    false, afr[i], false, bfr, (short)0, acc[i][j], false, false);
            }
        }
        __syncthreads();           // done reading buf 'cur' before it is refilled
        cur ^= 1;
    }

    // ---- write back: C/D layout: VGPR v -> M = v (+8 for lanes 16-31) ----
#pragma unroll
    for (int i = 0; i < 4; ++i) {
#pragma unroll
        for (int j = 0; j < 4; ++j) {
            const int col     = n0 + wc * 64 + j * 16 + (lane & 15);
            const int rowBase = m0 + wr * 64 + i * 16 + ((lane < 16) ? 0 : 8);
#pragma unroll
            for (int v = 0; v < 8; ++v) {
                C[(size_t)(rowBase + v) * N + col] = acc[i][j][v];
            }
        }
    }
}

// ---------------------------------------------------------------------------
// h_next[r,:] = silu( LN( tmp1[r,:] + tmp2[r/2,:] ) * g + b )
// one 256-thread block per row (D=1024 -> float4 per thread)
// ---------------------------------------------------------------------------
__global__ __launch_bounds__(256) void ln_silu_kernel(
    const float* __restrict__ t1, const float* __restrict__ t2,
    const float* __restrict__ g, const float* __restrict__ b,
    float* __restrict__ hout) {
    __shared__ float s1[256];
    __shared__ float s2[256];
    const int r   = blockIdx.x;
    const int tid = threadIdx.x;

    const float4 v1 = ((const float4*)(t1 + (size_t)r * D_DIM))[tid];
    const float4 v2 = ((const float4*)(t2 + (size_t)(r >> 1) * D_DIM))[tid];
    float4 v;
    v.x = v1.x + v2.x; v.y = v1.y + v2.y; v.z = v1.z + v2.z; v.w = v1.w + v2.w;

    s1[tid] = v.x + v.y + v.z + v.w;
    s2[tid] = v.x * v.x + v.y * v.y + v.z * v.z + v.w * v.w;
    __syncthreads();
#pragma unroll
    for (int off = 128; off > 0; off >>= 1) {
        if (tid < off) { s1[tid] += s1[tid + off]; s2[tid] += s2[tid + off]; }
        __syncthreads();
    }
    const float mu  = s1[0] * (1.0f / D_DIM);
    const float var = s2[0] * (1.0f / D_DIM) - mu * mu;
    const float inv = rsqrtf(var + 1e-5f);

    const float4 g4 = ((const float4*)g)[tid];
    const float4 b4 = ((const float4*)b)[tid];
    float4 o;
    o.x = silu_f((v.x - mu) * inv * g4.x + b4.x);
    o.y = silu_f((v.y - mu) * inv * g4.y + b4.y);
    o.z = silu_f((v.z - mu) * inv * g4.z + b4.z);
    o.w = silu_f((v.w - mu) * inv * g4.w + b4.w);
    ((float4*)(hout + (size_t)r * D_DIM))[tid] = o;
}

// ---------------------------------------------------------------------------
// out[b] = mean_s( h[b,s,:] . out_W ) + out_b   (one block per batch element)
// ---------------------------------------------------------------------------
__global__ __launch_bounds__(256) void pool_out_kernel(
    const float* __restrict__ h, const float* __restrict__ w,
    const float* __restrict__ ob, float* __restrict__ out) {
    __shared__ float s[256];
    const int bIdx = blockIdx.x;
    const int tid  = threadIdx.x;
    const float* hb = h + (size_t)bIdx * S_DIM * D_DIM;
    float acc = 0.0f;
    for (int idx = tid; idx < S_DIM * D_DIM; idx += 256) {
        acc += hb[idx] * w[idx & (D_DIM - 1)];
    }
    s[tid] = acc;
    __syncthreads();
#pragma unroll
    for (int off = 128; off > 0; off >>= 1) {
        if (tid < off) s[tid] += s[tid + off];
        __syncthreads();
    }
    if (tid == 0) out[bIdx] = s[0] * (1.0f / S_DIM) + ob[0];
}

// ---------------------------------------------------------------------------
extern "C" void kernel_launch(void* const* d_in, const int* in_sizes, int n_in,
                              void* d_out, int out_size, void* d_ws, size_t ws_size,
                              hipStream_t stream) {
    (void)in_sizes; (void)n_in; (void)out_size; (void)ws_size;

    const float* x      = (const float*)d_in[0];
    const float* base_W = (const float*)d_in[1];
    const float* wav_W  = (const float*)d_in[2];
    const float* ln_g   = (const float*)d_in[3];
    const float* ln_b   = (const float*)d_in[4];
    const float* out_W  = (const float*)d_in[5];
    const float* out_b  = (const float*)d_in[6];
    float* out = (float*)d_out;

    // ---- carve scratch out of d_ws ----
    char* p = (char*)d_ws;
    auto alloc = [&](size_t bytes) -> char* {
        char* r = p; p += (bytes + 255) & ~(size_t)255; return r;
    };
    unsigned short* baseWb = (unsigned short*)alloc((size_t)L_DIM * D_DIM * D_DIM * 2);
    unsigned short* wavWb  = (unsigned short*)alloc((size_t)L_DIM * D_DIM * 2 * D_DIM * 2);
    unsigned short* sA     = (unsigned short*)alloc((size_t)N_ROWS * D_DIM * 2);
    unsigned short* sP     = (unsigned short*)alloc((size_t)(N_ROWS / 2) * 2 * D_DIM * 2);
    float* tmp1 = (float*)alloc((size_t)N_ROWS * D_DIM * 4);
    float* tmp2 = (float*)alloc((size_t)(N_ROWS / 2) * D_DIM * 4);
    float* hbuf = (float*)alloc((size_t)N_ROWS * D_DIM * 4);

    // ---- convert weights to bf16 once per launch ----
    {
        int nbw = L_DIM * D_DIM * D_DIM;
        f32_to_bf16_kernel<<<(nbw + 255) / 256, 256, 0, stream>>>(base_W, baseWb, nbw);
        int nww = L_DIM * D_DIM * 2 * D_DIM;
        f32_to_bf16_kernel<<<(nww + 255) / 256, 256, 0, stream>>>(wav_W, wavWb, nww);
    }

    for (int i = 0; i < L_DIM; ++i) {
        const float* hin = (i == 0) ? x : hbuf;

        int nPrep = (N_ROWS / 2) * D_DIM;
        prep_silu_dwt_kernel<<<(nPrep + 255) / 256, 256, 0, stream>>>(
            hin, sA, sP, N_ROWS / 2);

        dim3 g1(D_DIM / 128, N_ROWS / 128);
        gemm_bf16_nt_kernel<<<g1, 128, 0, stream>>>(
            sA, baseWb + (size_t)i * D_DIM * D_DIM, tmp1, N_ROWS, D_DIM, D_DIM);

        dim3 g2(D_DIM / 128, (N_ROWS / 2) / 128);
        gemm_bf16_nt_kernel<<<g2, 128, 0, stream>>>(
            sP, wavWb + (size_t)i * D_DIM * 2 * D_DIM, tmp2, N_ROWS / 2, D_DIM, 2 * D_DIM);

        ln_silu_kernel<<<N_ROWS, 256, 0, stream>>>(
            tmp1, tmp2, ln_g + i * D_DIM, ln_b + i * D_DIM, hbuf);
    }

    pool_out_kernel<<<B_DIM, 256, 0, stream>>>(hbuf, out_W, out_b, out);
}
